// HybridPhysicsGNN_30743375905324
// MI455X (gfx1250) — matrix operand order repaired and checked by reading, not compiled, not run on hardware
//
#include <hip/hip_runtime.h>

// ---------------- constants ----------------
#define B_   2
#define N_   10000
#define E_   160000
#define D_   128
#define L_   4
#define DIN_ 5
#define NEG_ 0.2f
#define EPS_ 1e-5f

typedef __attribute__((ext_vector_type(16))) __bf16 bf16x16;
typedef __attribute__((ext_vector_type(8)))  __bf16 bf16x8;
typedef __attribute__((ext_vector_type(8)))  float  f32x8;

// ---------------- helpers ----------------
__device__ __forceinline__ unsigned short f2bf_bits(float f) {
  unsigned int u = __float_as_uint(f);
  unsigned int r = u + 0x7FFFu + ((u >> 16) & 1u);
  return (unsigned short)(r >> 16);
}

__device__ __forceinline__ float lrelu(float x) { return x > 0.f ? x : NEG_ * x; }

// sum across the 16-lane half-group (masks 1,2,4,8 stay within each half)
__device__ __forceinline__ float half16_sum(float x, int lane) {
#pragma unroll
  for (int mask = 1; mask <= 8; mask <<= 1) {
    int o = __builtin_amdgcn_ds_bpermute(((lane ^ mask) << 2), __float_as_int(x));
    x += __int_as_float(o);
  }
  return x;
}

// A-fragment (16x32 bf16) from row-major LDS tile.
// lane<16: row=lane,    K = kt*32 + {0..7, 16..23}
// lane>=16:row=lane-16, K = kt*32 + {8..15, 24..31}
__device__ __forceinline__ bf16x16 load_afrag(const unsigned short* base, int rowStride,
                                              int kt, int lane) {
  int m  = lane & 15;
  int hl = lane >> 4;
  const unsigned short* p = base + m * rowStride + kt * 32 + hl * 8;
  bf16x8 lo = *(const bf16x8*)p;
  bf16x8 hi = *(const bf16x8*)(p + 16);
  bf16x16 a;
#pragma unroll
  for (int i = 0; i < 8; ++i) { a[i] = lo[i]; a[i + 8] = hi[i]; }
  return a;
}

// B-fragment: pre-repacked so each lane's 16 elements are contiguous (32B load)
__device__ __forceinline__ bf16x16 load_bfrag(const unsigned short* fragBase,
                                              int blk, int lane) {
  return *(const bf16x16*)(fragBase + (size_t)blk * 512 + lane * 16);
}

// ---------------- weight fragment repack ----------------
// Repack W[K x Ncols] fp32 row-major into WMMA B-fragment order (bf16),
// zero-padding K beyond Ktrue. One wave per (kt,nt) block.
__global__ __launch_bounds__(32) void prep_frags(const float* __restrict__ W,
                                                 unsigned short* __restrict__ dst,
                                                 int Ktrue, int Ncols, int NT) {
  int blk = blockIdx.x;
  int kt = blk / NT, nt = blk % NT;
  int lane = threadIdx.x;
  int n = lane & 15, hl = lane >> 4;
  int kbase = kt * 32 + hl * 8;
  unsigned short* p = dst + (size_t)blk * 512 + lane * 16;
#pragma unroll
  for (int e = 0; e < 16; ++e) {
    int k = kbase + (e < 8 ? e : e + 8);
    float v = (k < Ktrue) ? W[(size_t)k * Ncols + nt * 16 + n] : 0.f;
    p[e] = f2bf_bits(v);
  }
}

// ---------------- encoder: h = x @ enc_W + enc_b ----------------
__global__ __launch_bounds__(128) void encoder_kernel(const float* __restrict__ x,
                                                      const float* __restrict__ W,
                                                      const float* __restrict__ bias,
                                                      float* __restrict__ h,
                                                      unsigned short* __restrict__ hbf) {
  size_t node = blockIdx.x;      // 0 .. B*N-1
  int d = threadIdx.x;           // 0 .. 127
  const float* xr = x + node * DIN_;
  float acc = bias[d];
#pragma unroll
  for (int i = 0; i < DIN_; ++i) acc = fmaf(xr[i], W[i * D_ + d], acc);
  h[node * D_ + d]   = acc;
  hbf[node * D_ + d] = f2bf_bits(acc);
}

// ---------------- edge message + scatter-add ----------------
#define EW 4  // waves per block; 20000 tiles / 4 = 5000 blocks exact
__global__ __launch_bounds__(32 * EW) void edge_msg_kernel(
    const unsigned short* __restrict__ hbf, const float* __restrict__ edge_attr,
    const int* __restrict__ edge_index, const unsigned short* __restrict__ fragW1,
    const unsigned short* __restrict__ fragW2, const float* __restrict__ b1,
    const float* __restrict__ lg, const float* __restrict__ lb,
    const float* __restrict__ b2, float* __restrict__ aggr, int layer) {
  __shared__ __align__(32) unsigned short ldsA[EW][16 * 288];
  __shared__ __align__(32) unsigned short ldsN[EW][16 * 128];
  __shared__ int ldsSrc[EW][16];

  int w = threadIdx.x >> 5, lane = threadIdx.x & 31;
  int m = lane & 15, hl = lane >> 4;
  int tile = blockIdx.x * EW + w;
  const int tilesPerBatch = E_ / 16;
  int bIdx = tile / tilesPerBatch;
  int e0 = (tile % tilesPerBatch) * 16;
  const int* rowI = edge_index;
  const int* colI = edge_index + E_;
  __builtin_prefetch(rowI + e0, 0, 1);

  // ---- stage A1 = [h[row] | h[col] | ea | 0-pad] in LDS (16 x 288 bf16) ----
#pragma unroll 4
  for (int r = 0; r < 16; ++r) {
    int e = e0 + r;
    int srcn = rowI[e], dstn = colI[e];
    if (lane == 0) ldsSrc[w][r] = srcn;
    int node = hl ? dstn : srcn;
    const unsigned short* hp = hbf + ((size_t)bIdx * N_ + node) * D_ + m * 8;
    *(bf16x8*)&ldsA[w][r * 288 + hl * 128 + m * 8] = *(const bf16x8*)hp;
  }
  {
    int r = m;
    unsigned short* p = &ldsA[w][r * 288 + 256 + hl * 16];
#pragma unroll
    for (int i = 0; i < 16; ++i) p[i] = 0;
    if (hl == 0) {
      int e = e0 + r;
      p[0] = f2bf_bits(edge_attr[(size_t)e * 2 + 0]);
      p[1] = f2bf_bits(edge_attr[(size_t)e * 2 + 1]);
    }
  }
  __syncthreads();

  // per-column constants (col = t*16 + m)
  float b1v[8], gv[8], bv[8], b2v[8];
#pragma unroll
  for (int t = 0; t < 8; ++t) {
    int c = layer * D_ + t * 16 + m;
    b1v[t] = b1[c]; gv[t] = lg[c]; bv[t] = lb[c]; b2v[t] = b2[c];
  }

  // ---- GEMM1: [16x288] @ W1[288x128] ----
  f32x8 acc[8];
#pragma unroll
  for (int t = 0; t < 8; ++t)
#pragma unroll
    for (int v = 0; v < 8; ++v) acc[t][v] = 0.f;

  const unsigned short* w1p = fragW1 + (size_t)layer * 72 * 512;
#pragma unroll
  for (int kt = 0; kt < 9; ++kt) {
    bf16x16 a = load_afrag(&ldsA[w][0], 288, kt, lane);
#pragma unroll
    for (int t = 0; t < 8; ++t) {
      bf16x16 bm = load_bfrag(w1p, kt * 8 + t, lane);
      acc[t] = __builtin_amdgcn_wmma_f32_16x16x32_bf16(false, a, false, bm,
                                                       (short)0, acc[t], false, false);
    }
  }

  // ---- +b1, LeakyReLU, LayerNorm (per-row over 128 cols) ----
  float mu[8], rs[8];
#pragma unroll
  for (int v = 0; v < 8; ++v) {
    float s = 0.f, s2 = 0.f;
#pragma unroll
    for (int t = 0; t < 8; ++t) {
      float xv = lrelu(acc[t][v] + b1v[t]);
      acc[t][v] = xv;
      s += xv; s2 = fmaf(xv, xv, s2);
    }
    s = half16_sum(s, lane);
    s2 = half16_sum(s2, lane);
    float mean = s * (1.f / 128.f);
    float var = s2 * (1.f / 128.f) - mean * mean;
    mu[v] = mean;
    rs[v] = rsqrtf(var + EPS_);
  }
#pragma unroll
  for (int t = 0; t < 8; ++t)
#pragma unroll
    for (int v = 0; v < 8; ++v) {
      float y = (acc[t][v] - mu[v]) * rs[v] * gv[t] + bv[t];
      int row = v + 8 * hl;
      ldsN[w][row * 128 + t * 16 + m] = f2bf_bits(y);
    }
  __syncthreads();

  // ---- GEMM2: [16x128] @ W2[128x128] ----
  f32x8 acc2[8];
#pragma unroll
  for (int t = 0; t < 8; ++t)
#pragma unroll
    for (int v = 0; v < 8; ++v) acc2[t][v] = 0.f;
  const unsigned short* w2p = fragW2 + (size_t)layer * 32 * 512;
#pragma unroll
  for (int kt = 0; kt < 4; ++kt) {
    bf16x16 a = load_afrag(&ldsN[w][0], 128, kt, lane);
#pragma unroll
    for (int t = 0; t < 8; ++t) {
      bf16x16 bm = load_bfrag(w2p, kt * 8 + t, lane);
      acc2[t] = __builtin_amdgcn_wmma_f32_16x16x32_bf16(false, a, false, bm,
                                                        (short)0, acc2[t], false, false);
    }
  }

  // ---- scatter-add onto source nodes ----
#pragma unroll
  for (int v = 0; v < 8; ++v) {
    int row = v + 8 * hl;
    int srcn = ldsSrc[w][row];
    float* ap = aggr + ((size_t)bIdx * N_ + srcn) * D_;
#pragma unroll
    for (int t = 0; t < 8; ++t)
      atomicAdd(ap + t * 16 + m, acc2[t][v] + b2v[t]);
  }
}

// ---------------- node update ----------------
#define NW 2  // 1250 tiles / 2 = 625 blocks exact
__global__ __launch_bounds__(32 * NW) void node_update_kernel(
    const unsigned short* __restrict__ hbf, const float* __restrict__ aggr,
    const unsigned short* __restrict__ fragUW, const float* __restrict__ ub,
    const float* __restrict__ ulg, const float* __restrict__ ulb,
    float* __restrict__ h, unsigned short* __restrict__ hbf_out, int layer) {
  __shared__ __align__(32) unsigned short ldsA[NW][16 * 256];

  int w = threadIdx.x >> 5, lane = threadIdx.x & 31;
  int m = lane & 15, hl = lane >> 4;
  int tile = blockIdx.x * NW + w;
  const int tilesPerBatch = N_ / 16;
  int bIdx = tile / tilesPerBatch;
  int n0 = (tile % tilesPerBatch) * 16;

  // stage [h | aggr] (16 x 256 bf16)
#pragma unroll 4
  for (int r = 0; r < 16; ++r) {
    size_t base = ((size_t)bIdx * N_ + n0 + r) * D_ + m * 8;
    if (hl == 0) {
      *(bf16x8*)&ldsA[w][r * 256 + m * 8] = *(const bf16x8*)(hbf + base);
    } else {
      const float* ap = aggr + base;
      unsigned short* p = &ldsA[w][r * 256 + 128 + m * 8];
#pragma unroll
      for (int i = 0; i < 8; ++i) p[i] = f2bf_bits(ap[i]);
    }
  }
  __syncthreads();

  float ubv[8], gv[8], bv[8];
#pragma unroll
  for (int t = 0; t < 8; ++t) {
    int c = layer * D_ + t * 16 + m;
    ubv[t] = ub[c]; gv[t] = ulg[c]; bv[t] = ulb[c];
  }

  f32x8 acc[8];
#pragma unroll
  for (int t = 0; t < 8; ++t)
#pragma unroll
    for (int v = 0; v < 8; ++v) acc[t][v] = 0.f;
  const unsigned short* uwp = fragUW + (size_t)layer * 64 * 512;
#pragma unroll
  for (int kt = 0; kt < 8; ++kt) {
    bf16x16 a = load_afrag(&ldsA[w][0], 256, kt, lane);
#pragma unroll
    for (int t = 0; t < 8; ++t) {
      bf16x16 bm = load_bfrag(uwp, kt * 8 + t, lane);
      acc[t] = __builtin_amdgcn_wmma_f32_16x16x32_bf16(false, a, false, bm,
                                                       (short)0, acc[t], false, false);
    }
  }

  float mu[8], rs[8];
#pragma unroll
  for (int v = 0; v < 8; ++v) {
    float s = 0.f, s2 = 0.f;
#pragma unroll
    for (int t = 0; t < 8; ++t) {
      float xv = lrelu(acc[t][v] + ubv[t]);
      acc[t][v] = xv;
      s += xv; s2 = fmaf(xv, xv, s2);
    }
    s = half16_sum(s, lane);
    s2 = half16_sum(s2, lane);
    float mean = s * (1.f / 128.f);
    float var = s2 * (1.f / 128.f) - mean * mean;
    mu[v] = mean;
    rs[v] = rsqrtf(var + EPS_);
  }

  // residual: h += LN(...)
#pragma unroll
  for (int v = 0; v < 8; ++v) {
    int row = v + 8 * hl;
    size_t base = ((size_t)bIdx * N_ + n0 + row) * D_;
#pragma unroll
    for (int t = 0; t < 8; ++t) {
      float y = (acc[t][v] - mu[v]) * rs[v] * gv[t] + bv[t];
      size_t idx = base + t * 16 + m;
      float hn = h[idx] + y;
      h[idx] = hn;
      hbf_out[idx] = f2bf_bits(hn);
    }
  }
}

// ---------------- global context: partial sums of lrelu(h@Wa+ba)@Wb ----------------
__global__ __launch_bounds__(32 * NW) void gc_partial_kernel(
    const unsigned short* __restrict__ hbf, const unsigned short* __restrict__ fragWa,
    const unsigned short* __restrict__ fragWb, const float* __restrict__ ba,
    float* __restrict__ gsum, int layer) {
  __shared__ __align__(32) unsigned short ldsA[NW][16 * 128];
  __shared__ __align__(32) unsigned short ldsN[NW][16 * 128];

  int w = threadIdx.x >> 5, lane = threadIdx.x & 31;
  int m = lane & 15, hl = lane >> 4;
  int tile = blockIdx.x * NW + w;
  const int tilesPerBatch = N_ / 16;
  int bIdx = tile / tilesPerBatch;
  int n0 = (tile % tilesPerBatch) * 16;

#pragma unroll
  for (int it = 0; it < 8; ++it) {
    int r = it * 2 + hl;
    size_t base = ((size_t)bIdx * N_ + n0 + r) * D_ + m * 8;
    *(bf16x8*)&ldsA[w][r * 128 + m * 8] = *(const bf16x8*)(hbf + base);
  }
  __syncthreads();

  float bav[8];
#pragma unroll
  for (int t = 0; t < 8; ++t) bav[t] = ba[layer * D_ + t * 16 + m];

  f32x8 acc[8];
#pragma unroll
  for (int t = 0; t < 8; ++t)
#pragma unroll
    for (int v = 0; v < 8; ++v) acc[t][v] = 0.f;
  const unsigned short* wap = fragWa + (size_t)layer * 32 * 512;
#pragma unroll
  for (int kt = 0; kt < 4; ++kt) {
    bf16x16 a = load_afrag(&ldsA[w][0], 128, kt, lane);
#pragma unroll
    for (int t = 0; t < 8; ++t) {
      bf16x16 bm = load_bfrag(wap, kt * 8 + t, lane);
      acc[t] = __builtin_amdgcn_wmma_f32_16x16x32_bf16(false, a, false, bm,
                                                       (short)0, acc[t], false, false);
    }
  }
#pragma unroll
  for (int t = 0; t < 8; ++t)
#pragma unroll
    for (int v = 0; v < 8; ++v) {
      int row = v + 8 * hl;
      ldsN[w][row * 128 + t * 16 + m] = f2bf_bits(lrelu(acc[t][v] + bav[t]));
    }
  __syncthreads();

  f32x8 acc2[8];
#pragma unroll
  for (int t = 0; t < 8; ++t)
#pragma unroll
    for (int v = 0; v < 8; ++v) acc2[t][v] = 0.f;
  const unsigned short* wbp = fragWb + (size_t)layer * 32 * 512;
#pragma unroll
  for (int kt = 0; kt < 4; ++kt) {
    bf16x16 a = load_afrag(&ldsN[w][0], 128, kt, lane);
#pragma unroll
    for (int t = 0; t < 8; ++t) {
      bf16x16 bm = load_bfrag(wbp, kt * 8 + t, lane);
      acc2[t] = __builtin_amdgcn_wmma_f32_16x16x32_bf16(false, a, false, bm,
                                                        (short)0, acc2[t], false, false);
    }
  }
  // local row-sum then atomic partial sum per column (bb added later in gate kernel)
#pragma unroll
  for (int t = 0; t < 8; ++t) {
    float s = 0.f;
#pragma unroll
    for (int v = 0; v < 8; ++v) s += acc2[t][v];
    atomicAdd(gsum + bIdx * D_ + t * 16 + m, s);
  }
}

// ---------------- gate = sigmoid((gsum/N + bb) @ Wc + bc) ----------------
__global__ __launch_bounds__(128) void gate_kernel(const float* __restrict__ gsum,
                                                   const float* __restrict__ bb,
                                                   const float* __restrict__ Wc,
                                                   const float* __restrict__ bc,
                                                   float* __restrict__ gate, int layer) {
  int bIdx = blockIdx.x, d = threadIdx.x;
  __shared__ float gs[D_];
  gs[d] = gsum[bIdx * D_ + d] * (1.f / (float)N_) + bb[layer * D_ + d];
  __syncthreads();
  float a = bc[layer * D_ + d];
  const float* wc = Wc + (size_t)layer * D_ * D_;
#pragma unroll 4
  for (int k = 0; k < D_; ++k) a = fmaf(gs[k], wc[k * D_ + d], a);
  gate[bIdx * D_ + d] = 1.f / (1.f + __expf(-a));
}

// ---------------- apply gate: h *= gate; refresh bf16 shadow ----------------
__global__ __launch_bounds__(128) void apply_gate_kernel(float* __restrict__ h,
                                                         unsigned short* __restrict__ hbf,
                                                         const float* __restrict__ gate) {
  size_t node = blockIdx.x;            // 0..B*N-1
  int d = threadIdx.x;
  int bIdx = (int)(node / N_);
  size_t idx = node * D_ + d;
  float v = h[idx] * gate[bIdx * D_ + d];
  h[idx] = v;
  hbf[idx] = f2bf_bits(v);
}

// ---------------- decoder ----------------
__global__ __launch_bounds__(32 * NW) void decoder_kernel(
    const unsigned short* __restrict__ hbf, const unsigned short* __restrict__ fragD1,
    const float* __restrict__ db1, const float* __restrict__ W2,
    const float* __restrict__ db2, float* __restrict__ out) {
  __shared__ __align__(32) unsigned short ldsA[NW][16 * 128];
  __shared__ __align__(16) float ldsD[NW][16 * 64];

  int w = threadIdx.x >> 5, lane = threadIdx.x & 31;
  int m = lane & 15, hl = lane >> 4;
  int tile = blockIdx.x * NW + w;
  const int tilesPerBatch = N_ / 16;
  int bIdx = tile / tilesPerBatch;
  int n0 = (tile % tilesPerBatch) * 16;

#pragma unroll
  for (int it = 0; it < 8; ++it) {
    int r = it * 2 + hl;
    size_t base = ((size_t)bIdx * N_ + n0 + r) * D_ + m * 8;
    *(bf16x8*)&ldsA[w][r * 128 + m * 8] = *(const bf16x8*)(hbf + base);
  }
  __syncthreads();

  float b1v[4];
#pragma unroll
  for (int t = 0; t < 4; ++t) b1v[t] = db1[t * 16 + m];

  f32x8 acc[4];
#pragma unroll
  for (int t = 0; t < 4; ++t)
#pragma unroll
    for (int v = 0; v < 8; ++v) acc[t][v] = 0.f;
#pragma unroll
  for (int kt = 0; kt < 4; ++kt) {
    bf16x16 a = load_afrag(&ldsA[w][0], 128, kt, lane);
#pragma unroll
    for (int t = 0; t < 4; ++t) {
      bf16x16 bm = load_bfrag(fragD1, kt * 4 + t, lane);
      acc[t] = __builtin_amdgcn_wmma_f32_16x16x32_bf16(false, a, false, bm,
                                                       (short)0, acc[t], false, false);
    }
  }
#pragma unroll
  for (int t = 0; t < 4; ++t)
#pragma unroll
    for (int v = 0; v < 8; ++v) {
      int row = v + 8 * hl;
      ldsD[w][row * 64 + t * 16 + m] = lrelu(acc[t][v] + b1v[t]);
    }
  __syncthreads();

  // tail GEMM [16x64] @ [64x2] : one output per lane
  int row = m, ch = hl;
  float s = db2[ch];
#pragma unroll 8
  for (int k = 0; k < 64; ++k) s = fmaf(ldsD[w][row * 64 + k], W2[k * 2 + ch], s);
  out[((size_t)bIdx * N_ + n0 + row) * 2 + ch] = s;
}

// ---------------- launch ----------------
extern "C" void kernel_launch(void* const* d_in, const int* in_sizes, int n_in,
                              void* d_out, int out_size, void* d_ws, size_t ws_size,
                              hipStream_t stream) {
  const float* x_batch   = (const float*)d_in[0];
  const float* edge_attr = (const float*)d_in[1];
  const int*   edge_idx  = (const int*)d_in[2];
  const float* enc_W  = (const float*)d_in[3];
  const float* enc_b  = (const float*)d_in[4];
  const float* msg_W1 = (const float*)d_in[5];
  const float* msg_b1 = (const float*)d_in[6];
  const float* msg_lg = (const float*)d_in[7];
  const float* msg_lb = (const float*)d_in[8];
  const float* msg_W2 = (const float*)d_in[9];
  const float* msg_b2 = (const float*)d_in[10];
  const float* up_W   = (const float*)d_in[11];
  const float* up_b   = (const float*)d_in[12];
  const float* up_lg  = (const float*)d_in[13];
  const float* up_lb  = (const float*)d_in[14];
  const float* gc_Wa  = (const float*)d_in[15];
  const float* gc_ba  = (const float*)d_in[16];
  const float* gc_Wb  = (const float*)d_in[17];
  const float* gc_bb  = (const float*)d_in[18];
  const float* gc_Wc  = (const float*)d_in[19];
  const float* gc_bc  = (const float*)d_in[20];
  const float* dec_W1 = (const float*)d_in[21];
  const float* dec_b1 = (const float*)d_in[22];
  const float* dec_W2 = (const float*)d_in[23];
  const float* dec_b2 = (const float*)d_in[24];
  float* out = (float*)d_out;

  char* ws = (char*)d_ws;
  size_t off = 0;
  auto carve = [&](size_t bytes) {
    char* p = ws + off;
    off = (off + bytes + 255) & ~(size_t)255;
    return p;
  };
  float*          h     = (float*)carve((size_t)B_ * N_ * D_ * 4);
  unsigned short* hbf   = (unsigned short*)carve((size_t)B_ * N_ * D_ * 2);
  float*          aggr  = (float*)carve((size_t)B_ * N_ * D_ * 4);
  float*          gsum  = (float*)carve((size_t)B_ * D_ * 4);
  float*          gate  = (float*)carve((size_t)B_ * D_ * 4);
  unsigned short* fW1   = (unsigned short*)carve((size_t)L_ * 72 * 512 * 2);
  unsigned short* fW2   = (unsigned short*)carve((size_t)L_ * 32 * 512 * 2);
  unsigned short* fUW   = (unsigned short*)carve((size_t)L_ * 64 * 512 * 2);
  unsigned short* fWa   = (unsigned short*)carve((size_t)L_ * 32 * 512 * 2);
  unsigned short* fWb   = (unsigned short*)carve((size_t)L_ * 32 * 512 * 2);
  unsigned short* fD1   = (unsigned short*)carve((size_t)16 * 512 * 2);

  // --- weight repack ---
  for (int l = 0; l < L_; ++l) {
    prep_frags<<<72, 32, 0, stream>>>(msg_W1 + (size_t)l * 258 * 128, fW1 + (size_t)l * 72 * 512, 258, 128, 8);
    prep_frags<<<32, 32, 0, stream>>>(msg_W2 + (size_t)l * 128 * 128, fW2 + (size_t)l * 32 * 512, 128, 128, 8);
    prep_frags<<<64, 32, 0, stream>>>(up_W  + (size_t)l * 256 * 128, fUW + (size_t)l * 64 * 512, 256, 128, 8);
    prep_frags<<<32, 32, 0, stream>>>(gc_Wa + (size_t)l * 128 * 128, fWa + (size_t)l * 32 * 512, 128, 128, 8);
    prep_frags<<<32, 32, 0, stream>>>(gc_Wb + (size_t)l * 128 * 128, fWb + (size_t)l * 32 * 512, 128, 128, 8);
  }
  prep_frags<<<16, 32, 0, stream>>>(dec_W1, fD1, 128, 64, 4);

  // --- encoder ---
  encoder_kernel<<<B_ * N_, 128, 0, stream>>>(x_batch, enc_W, enc_b, h, hbf);

  // --- layers ---
  const int edgeBlocks = (B_ * E_ / 16) / EW;  // 5000
  const int nodeBlocks = (B_ * N_ / 16) / NW;  // 625
  for (int l = 0; l < L_; ++l) {
    hipMemsetAsync(aggr, 0, (size_t)B_ * N_ * D_ * 4, stream);
    edge_msg_kernel<<<edgeBlocks, 32 * EW, 0, stream>>>(
        hbf, edge_attr, edge_idx, fW1, fW2, msg_b1, msg_lg, msg_lb, msg_b2, aggr, l);
    node_update_kernel<<<nodeBlocks, 32 * NW, 0, stream>>>(
        hbf, aggr, fUW, up_b, up_lg, up_lb, h, hbf, l);
    hipMemsetAsync(gsum, 0, (size_t)B_ * D_ * 4, stream);
    gc_partial_kernel<<<nodeBlocks, 32 * NW, 0, stream>>>(hbf, fWa, fWb, gc_ba, gsum, l);
    gate_kernel<<<B_, 128, 0, stream>>>(gsum, gc_bb, gc_Wc, gc_bc, gate, l);
    apply_gate_kernel<<<B_ * N_, 128, 0, stream>>>(h, hbf, gate);
  }

  // --- decoder ---
  decoder_kernel<<<nodeBlocks, 32 * NW, 0, stream>>>(hbf, fD1, dec_b1, dec_W2, dec_b2, out);
}
